// TwosDiacEncoder_21775484191106
// MI455X (gfx1250) — matrix operand order repaired and checked by reading, not compiled, not run on hardware
//
#include <hip/hip_runtime.h>

// ---------------------------------------------------------------------------
// CDNA5 / gfx1250 transformer encoder block (wave32, WMMA bf16 -> f32 accum)
// GEMMs: 32x64 tile/wave, 8 v_wmma per k-step. Attention: flash-style with
// double-buffered GLOBAL_LOAD_ASYNC_TO_LDS_B128 K/V staging (ASYNCcnt).
// ---------------------------------------------------------------------------

typedef __attribute__((ext_vector_type(16))) __bf16 v16bf;
typedef __attribute__((ext_vector_type(8)))  float  v8f;

#define B_   16
#define T_   1024
#define D_   512
#define H_   4
#define KC_  128
#define FF_  1024
#define M_   (B_ * T_)
#define EPS_ 1e-5f

// --- async global->LDS support (guarded; falls back to direct global loads) -
#if __has_builtin(__builtin_amdgcn_global_load_async_to_lds_b128)
#define HAVE_ASYNC_LDS 1
typedef __attribute__((__vector_size__(4 * sizeof(int)))) int i4v;
typedef __attribute__((address_space(1))) i4v* gas_i4v_p;   // global int4*
typedef __attribute__((address_space(3))) i4v* las_i4v_p;   // LDS int4*
#define ASYNC_CP16(gsrc, ldst) \
  __builtin_amdgcn_global_load_async_to_lds_b128((gas_i4v_p)(gsrc), (las_i4v_p)(ldst), 0, 0)
#if __has_builtin(__builtin_amdgcn_s_wait_asynccnt)
#define WAIT_ASYNC0() __builtin_amdgcn_s_wait_asynccnt(0)
#else
#define WAIT_ASYNC0() asm volatile("s_wait_asynccnt 0x0" ::: "memory")
#endif
#endif

__device__ __forceinline__ unsigned short f2bf(float f) {
  union { float f; unsigned int u; } x; x.f = f;
  unsigned int u = x.u;
  return (unsigned short)((u + 0x7FFFu + ((u >> 16) & 1u)) >> 16);  // RNE
}

union FragU { v16bf v; uint4 q[2]; unsigned short u[16]; };

// ----------------------------- elementwise casts ---------------------------

__global__ void __launch_bounds__(256)
cast_bf16_kernel(const float* __restrict__ in, unsigned short* __restrict__ out, int n) {
  int i = blockIdx.x * blockDim.x + threadIdx.x;
  if (i < n) out[i] = f2bf(in[i]);
}

// v [B,T,D] (head-major cols) -> vt [B,H,KC,T] so P*V B-fragments are linear
__global__ void __launch_bounds__(256)
transpose_v_kernel(const unsigned short* __restrict__ v, unsigned short* __restrict__ vt) {
  int i = blockIdx.x * blockDim.x + threadIdx.x;   // over B*H*KC*T
  int s  = i & (T_ - 1);
  int r  = i >> 10;
  int c  = r & (KC_ - 1);
  int r2 = r >> 7;
  int h  = r2 & (H_ - 1);
  int b  = r2 >> 2;
  vt[i] = v[((size_t)(b * T_ + s)) * D_ + h * KC_ + c];
}

// ----------------------------- layernorm (+mask, ->bf16) -------------------

__global__ void __launch_bounds__(128)
ln_kernel(const float* __restrict__ in, const float* __restrict__ gw,
          const float* __restrict__ bw, const int* __restrict__ lengths,
          unsigned short* __restrict__ out) {
  int row = blockIdx.x;
  int tid = threadIdx.x;                 // 128 threads, D=512 -> 4 elems each
  const float* x = in + (size_t)row * D_;
  float v0 = x[tid], v1 = x[tid + 128], v2 = x[tid + 256], v3 = x[tid + 384];
  float s  = v0 + v1 + v2 + v3;
  float s2 = v0 * v0 + v1 * v1 + v2 * v2 + v3 * v3;
#pragma unroll
  for (int off = 16; off; off >>= 1) {
    s  += __shfl_xor(s, off, 32);
    s2 += __shfl_xor(s2, off, 32);
  }
  __shared__ float sb[8];
  int w = tid >> 5;
  if ((tid & 31) == 0) { sb[w] = s; sb[4 + w] = s2; }
  __syncthreads();
  s  = sb[0] + sb[1] + sb[2] + sb[3];
  s2 = sb[4] + sb[5] + sb[6] + sb[7];
  float mean = s * (1.0f / D_);
  float var  = s2 * (1.0f / D_) - mean * mean;
  float rstd = rsqrtf(var + EPS_);
  float mk = 1.0f;
  if (lengths) {
    int b = row / T_, t = row % T_;
    mk = (t < lengths[b]) ? 1.0f : 0.0f;
  }
  unsigned short* y = out + (size_t)row * D_;
#pragma unroll
  for (int i = 0; i < 4; ++i) {
    int c = tid + i * 128;
    y[c] = f2bf(((x[c] - mean) * rstd * gw[c] + bw[c]) * mk);
  }
}

// ----------------------------- WMMA GEMM -----------------------------------
// Y[M,N] = act(X[M,K](bf16) @ W[N,K]^T(bf16) + bias) (+resid) ; out f32/bf16.
// One wave owns a 32x64 tile: 2 A-frags x 4 B-frags -> 8 v_wmma per k-step.

__global__ void __launch_bounds__(128)
gemm_wmma_kernel(const unsigned short* __restrict__ X,
                 const unsigned short* __restrict__ W,
                 const float* __restrict__ bias,
                 const float* __restrict__ resid,
                 float* __restrict__ outF,
                 unsigned short* __restrict__ outB,
                 int M, int N, int K, int relu) {
  int wid  = (int)((blockIdx.x * blockDim.x + threadIdx.x) >> 5);
  int lane = threadIdx.x & 31;
  int tiles_n = N >> 6;
  int m0 = (wid / tiles_n) << 5;
  int n0 = (wid % tiles_n) << 6;
  int half = lane >> 4;
  int l15  = lane & 15;

  const unsigned short* xr0 = X + (size_t)(m0 + l15) * K + half * 16;       // A rows
  const unsigned short* xr1 = xr0 + (size_t)16 * K;
  const unsigned short* wr  = W + (size_t)(n0 + l15) * K + half * 16;       // B rows
  size_t ws = (size_t)16 * K;

  v8f acc[2][4];
#pragma unroll
  for (int i = 0; i < 2; ++i)
#pragma unroll
    for (int j = 0; j < 4; ++j) acc[i][j] = (v8f){};

  for (int k0 = 0; k0 < K; k0 += 32) {
    FragU a0, a1;
    a0.q[0] = *(const uint4*)(xr0 + k0);
    a0.q[1] = *(const uint4*)(xr0 + k0 + 8);
    a1.q[0] = *(const uint4*)(xr1 + k0);
    a1.q[1] = *(const uint4*)(xr1 + k0 + 8);
    __builtin_prefetch(xr0 + k0 + 64, 0, 0);   // global_prefetch_b8
    FragU b0, b1, b2, b3;
    const unsigned short* w0 = wr + k0;
    b0.q[0] = *(const uint4*)(w0);           b0.q[1] = *(const uint4*)(w0 + 8);
    b1.q[0] = *(const uint4*)(w0 + ws);      b1.q[1] = *(const uint4*)(w0 + ws + 8);
    b2.q[0] = *(const uint4*)(w0 + 2 * ws);  b2.q[1] = *(const uint4*)(w0 + 2 * ws + 8);
    b3.q[0] = *(const uint4*)(w0 + 3 * ws);  b3.q[1] = *(const uint4*)(w0 + 3 * ws + 8);
    acc[0][0] = __builtin_amdgcn_wmma_f32_16x16x32_bf16(false, a0.v, false, b0.v, (short)0, acc[0][0], false, false);
    acc[0][1] = __builtin_amdgcn_wmma_f32_16x16x32_bf16(false, a0.v, false, b1.v, (short)0, acc[0][1], false, false);
    acc[0][2] = __builtin_amdgcn_wmma_f32_16x16x32_bf16(false, a0.v, false, b2.v, (short)0, acc[0][2], false, false);
    acc[0][3] = __builtin_amdgcn_wmma_f32_16x16x32_bf16(false, a0.v, false, b3.v, (short)0, acc[0][3], false, false);
    acc[1][0] = __builtin_amdgcn_wmma_f32_16x16x32_bf16(false, a1.v, false, b0.v, (short)0, acc[1][0], false, false);
    acc[1][1] = __builtin_amdgcn_wmma_f32_16x16x32_bf16(false, a1.v, false, b1.v, (short)0, acc[1][1], false, false);
    acc[1][2] = __builtin_amdgcn_wmma_f32_16x16x32_bf16(false, a1.v, false, b2.v, (short)0, acc[1][2], false, false);
    acc[1][3] = __builtin_amdgcn_wmma_f32_16x16x32_bf16(false, a1.v, false, b3.v, (short)0, acc[1][3], false, false);
  }

#pragma unroll
  for (int i = 0; i < 2; ++i) {
#pragma unroll
    for (int j = 0; j < 4; ++j) {
      int col = n0 + j * 16 + l15;
      float bv = bias ? bias[col] : 0.0f;
#pragma unroll
      for (int g = 0; g < 8; ++g) {
        int row = m0 + i * 16 + g + half * 8;
        float y = acc[i][j][g] + bv;
        if (relu) y = fmaxf(y, 0.0f);
        if (resid) y += resid[(size_t)row * N + col];
        if (outF) outF[(size_t)row * N + col] = y;
        if (outB) outB[(size_t)row * N + col] = f2bf(y);
      }
    }
  }
}

// ----------------------------- flash attention -----------------------------
// One wave per (b, h, 16-row q tile). S = QK^T/sqrt(128) - log1p|i-j|, masked,
// online softmax over 32-col chunks; O += P*V via WMMA after an LDS transpose.
// K/V tiles double-buffered in LDS via GLOBAL_LOAD_ASYNC_TO_LDS_B128.

__global__ void __launch_bounds__(32)
attn_kernel(const unsigned short* __restrict__ Q,
            const unsigned short* __restrict__ Km,
            const unsigned short* __restrict__ Vt,
            const int* __restrict__ lengths,
            unsigned short* __restrict__ O) {
#if defined(HAVE_ASYNC_LDS)
  __shared__ __align__(16) unsigned short sK[2][32 * 128];   // 2 x 8KB
  __shared__ __align__(16) unsigned short sV[2][128 * 32];   // 2 x 8KB
#endif
  __shared__ __align__(16) unsigned short lds_p[16 * 40];    // P transpose

  int bid = blockIdx.x;
  int tt = bid & 63;             // T/16 = 64
  int h  = (bid >> 6) & 3;
  int b  = bid >> 8;
  int t0 = tt << 4;
  int lane = threadIdx.x;
  int half = lane >> 4;
  int l15  = lane & 15;
  int len  = lengths[b];

  // Q A-fragments, K=128 -> 4 fragments of 16x32
  const unsigned short* qrow = Q + ((size_t)(b * T_ + t0 + l15)) * D_ + h * KC_ + half * 16;
  FragU qa[4];
#pragma unroll
  for (int kc = 0; kc < 4; ++kc) {
    qa[kc].q[0] = *(const uint4*)(qrow + kc * 32);
    qa[kc].q[1] = *(const uint4*)(qrow + kc * 32 + 8);
  }

  const unsigned short* kgbase = Km + ((size_t)b * T_) * D_ + h * KC_;   // [T,128]
  const unsigned short* vgbase = Vt + (((size_t)b * H_ + h) * KC_) * T_; // [128,T]

#if defined(HAVE_ASYNC_LDS)
  auto issue_tiles = [&](int bufI, int s0n) {
#pragma unroll
    for (int i = 0; i < 16; ++i) {
      int flat = i * 32 + lane;
      int kr = flat >> 4, kch = flat & 15;   // K tile: 32 rows x 16 chunks
      ASYNC_CP16(kgbase + (size_t)(s0n + kr) * D_ + kch * 8,
                 &sK[bufI][kr * 128 + kch * 8]);
      int vr = flat >> 2, vch = flat & 3;    // V tile: 128 rows x 4 chunks
      ASYNC_CP16(vgbase + (size_t)vr * T_ + s0n + vch * 8,
                 &sV[bufI][vr * 32 + vch * 8]);
    }
  };
  issue_tiles(0, 0);
  int buf = 0;
#else
  const unsigned short* kb = kgbase + half * 16;
  const unsigned short* vb = vgbase + half * 16;
#endif

  float mrow[8], lrow[8];
  v8f oacc[8];
#pragma unroll
  for (int g = 0; g < 8; ++g) { mrow[g] = -3.0e38f; lrow[g] = 0.0f; oacc[g] = (v8f){}; }

  const float scale = 0.08838834764831845f;  // 1/sqrt(128)

  for (int s0 = 0; s0 < T_; s0 += 32) {
#if defined(HAVE_ASYNC_LDS)
    WAIT_ASYNC0();                             // current buffer resident
    if (s0 + 32 < T_) issue_tiles(buf ^ 1, s0 + 32);  // overlap next with compute
#endif
    v8f sa0 = {}, sa1 = {};
#pragma unroll
    for (int kc = 0; kc < 4; ++kc) {
      FragU kf0, kf1;
#if defined(HAVE_ASYNC_LDS)
      const unsigned short* kp0 = &sK[buf][(l15) * 128 + kc * 32 + half * 16];
      const unsigned short* kp1 = &sK[buf][(16 + l15) * 128 + kc * 32 + half * 16];
#else
      const unsigned short* kp0 = kb + (size_t)(s0 + l15) * D_ + kc * 32;
      const unsigned short* kp1 = kb + (size_t)(s0 + 16 + l15) * D_ + kc * 32;
#endif
      kf0.q[0] = *(const uint4*)(kp0); kf0.q[1] = *(const uint4*)(kp0 + 8);
      kf1.q[0] = *(const uint4*)(kp1); kf1.q[1] = *(const uint4*)(kp1 + 8);
      sa0 = __builtin_amdgcn_wmma_f32_16x16x32_bf16(false, qa[kc].v, false, kf0.v, (short)0, sa0, false, false);
      sa1 = __builtin_amdgcn_wmma_f32_16x16x32_bf16(false, qa[kc].v, false, kf1.v, (short)0, sa1, false, false);
    }

    // bias + mask + online softmax (per-row stats live in the 16-lane group)
    float sv0[8], sv1[8], rmx[8];
#pragma unroll
    for (int g = 0; g < 8; ++g) {
      int irow = t0 + g + half * 8;
      int sc0 = s0 + l15;
      int sc1 = s0 + 16 + l15;
      float v0 = sa0[g] * scale - __logf(1.0f + fabsf((float)(irow - sc0)));
      float v1 = sa1[g] * scale - __logf(1.0f + fabsf((float)(irow - sc1)));
      bool mi = irow < len;
      v0 = (mi && sc0 < len) ? v0 : -1.0e4f;
      v1 = (mi && sc1 < len) ? v1 : -1.0e4f;
      sv0[g] = v0; sv1[g] = v1;
      float r = fmaxf(v0, v1);
#pragma unroll
      for (int off = 8; off; off >>= 1) r = fmaxf(r, __shfl_xor(r, off, 32));
      rmx[g] = r;
    }
#pragma unroll
    for (int g = 0; g < 8; ++g) {
      float mnew = fmaxf(mrow[g], rmx[g]);
      float corr = __expf(mrow[g] - mnew);
      float p0 = __expf(sv0[g] - mnew);
      float p1 = __expf(sv1[g] - mnew);
      float rs = p0 + p1;
#pragma unroll
      for (int off = 8; off; off >>= 1) rs += __shfl_xor(rs, off, 32);
      lrow[g] = lrow[g] * corr + rs;
      mrow[g] = mnew;
#pragma unroll
      for (int c = 0; c < 8; ++c) oacc[c][g] *= corr;
      int lr = g + half * 8;
      lds_p[lr * 40 + l15]      = f2bf(p0);
      lds_p[lr * 40 + 16 + l15] = f2bf(p1);
    }
    __syncthreads();

    // read P back in A-matrix layout (rows are 16B aligned: stride 40 ushorts)
    FragU pa;
    pa.q[0] = *(const uint4*)(lds_p + l15 * 40 + half * 16);
    pa.q[1] = *(const uint4*)(lds_p + l15 * 40 + 8 + half * 16);

#pragma unroll
    for (int c = 0; c < 8; ++c) {
      FragU vf;
#if defined(HAVE_ASYNC_LDS)
      const unsigned short* vp = &sV[buf][(c * 16 + l15) * 32 + half * 16];
#else
      const unsigned short* vp = vb + (size_t)(c * 16 + l15) * T_ + s0;
#endif
      vf.q[0] = *(const uint4*)(vp);
      vf.q[1] = *(const uint4*)(vp + 8);
      oacc[c] = __builtin_amdgcn_wmma_f32_16x16x32_bf16(false, pa.v, false, vf.v, (short)0, oacc[c], false, false);
    }
    __syncthreads();
#if defined(HAVE_ASYNC_LDS)
    buf ^= 1;
#endif
  }

  unsigned short* ob = O + ((size_t)(b * T_ + t0)) * D_ + h * KC_;
#pragma unroll
  for (int g = 0; g < 8; ++g) {
    float inv = 1.0f / lrow[g];
    int r = g + half * 8;
#pragma unroll
    for (int c = 0; c < 8; ++c) {
      ob[(size_t)r * D_ + c * 16 + l15] = f2bf(oacc[c][g] * inv);
    }
  }
}

// ----------------------------- launch --------------------------------------

extern "C" void kernel_launch(void* const* d_in, const int* in_sizes, int n_in,
                              void* d_out, int out_size, void* d_ws, size_t ws_size,
                              hipStream_t stream) {
  (void)in_sizes; (void)n_in; (void)out_size; (void)ws_size;

  const float* x     = (const float*)d_in[0];
  const float* z     = (const float*)d_in[1];
  const int*   lens  = (const int*)d_in[2];
  const float* ln1g  = (const float*)d_in[3];
  const float* ln1b  = (const float*)d_in[4];
  const float* wq    = (const float*)d_in[5];
  const float* bq    = (const float*)d_in[6];
  const float* wk    = (const float*)d_in[7];
  const float* bk    = (const float*)d_in[8];
  const float* wv    = (const float*)d_in[9];
  const float* bv    = (const float*)d_in[10];
  const float* wo    = (const float*)d_in[11];
  const float* bo    = (const float*)d_in[12];
  const float* ln2g  = (const float*)d_in[13];
  const float* ln2b  = (const float*)d_in[14];
  const float* w1    = (const float*)d_in[15];
  const float* b1    = (const float*)d_in[16];
  const float* w2    = (const float*)d_in[17];
  const float* b2    = (const float*)d_in[18];

  float* out      = (float*)d_out;
  float* attn_out = out + (size_t)M_ * D_;

  char* ws = (char*)d_ws;
  size_t off = 0;
  auto alloc = [&](size_t bytes) {
    char* p = ws + off;
    off += (bytes + 255) & ~(size_t)255;
    return p;
  };
  unsigned short* xm   = (unsigned short*)alloc((size_t)M_ * D_ * 2);
  unsigned short* qbuf = (unsigned short*)alloc((size_t)M_ * D_ * 2);
  unsigned short* kbuf = (unsigned short*)alloc((size_t)M_ * D_ * 2);
  unsigned short* vbuf = (unsigned short*)alloc((size_t)M_ * D_ * 2);
  unsigned short* vtb  = (unsigned short*)alloc((size_t)M_ * D_ * 2);
  unsigned short* obuf = (unsigned short*)alloc((size_t)M_ * D_ * 2);
  unsigned short* h2b  = (unsigned short*)alloc((size_t)M_ * D_ * 2);
  unsigned short* fbuf = (unsigned short*)alloc((size_t)M_ * FF_ * 2);
  unsigned short* wqb  = (unsigned short*)alloc((size_t)D_ * D_ * 2);
  unsigned short* wkb  = (unsigned short*)alloc((size_t)D_ * D_ * 2);
  unsigned short* wvb  = (unsigned short*)alloc((size_t)D_ * D_ * 2);
  unsigned short* wob  = (unsigned short*)alloc((size_t)D_ * D_ * 2);
  unsigned short* w1b  = (unsigned short*)alloc((size_t)FF_ * D_ * 2);
  unsigned short* w2b  = (unsigned short*)alloc((size_t)D_ * FF_ * 2);

  auto cast = [&](const float* src, unsigned short* dst, int n) {
    cast_bf16_kernel<<<(n + 255) / 256, 256, 0, stream>>>(src, dst, n);
  };
  cast(wq, wqb, D_ * D_);
  cast(wk, wkb, D_ * D_);
  cast(wv, wvb, D_ * D_);
  cast(wo, wob, D_ * D_);
  cast(w1, w1b, FF_ * D_);
  cast(w2, w2b, D_ * FF_);

  auto gemm = [&](const unsigned short* X, const unsigned short* W, const float* bias,
                  const float* resid, float* oF, unsigned short* oB,
                  int M, int N, int K, int relu) {
    int waves = (M / 32) * (N / 64);
    gemm_wmma_kernel<<<waves / 4, 128, 0, stream>>>(X, W, bias, resid, oF, oB, M, N, K, relu);
  };

  // LN1 + mask -> xm (bf16)
  ln_kernel<<<M_, 128, 0, stream>>>(x, ln1g, ln1b, lens, xm);

  // Q/K/V projections
  gemm(xm, wqb, bq, nullptr, nullptr, qbuf, M_, D_, D_, 0);
  gemm(xm, wkb, bk, nullptr, nullptr, kbuf, M_, D_, D_, 0);
  gemm(xm, wvb, bv, nullptr, nullptr, vbuf, M_, D_, D_, 0);

  // V -> [B,H,KC,T] for linear B-fragments in P*V
  transpose_v_kernel<<<(B_ * H_ * KC_ * T_) / 256, 256, 0, stream>>>(vbuf, vtb);

  // attention
  attn_kernel<<<B_ * H_ * (T_ / 16), 32, 0, stream>>>(qbuf, kbuf, vtb, lens, obuf);

  // output projection + z residual -> attn_out (f32, second output)
  gemm(obuf, wob, bo, z, attn_out, nullptr, M_, D_, D_, 0);

  // LN2 -> h (bf16)
  ln_kernel<<<M_, 128, 0, stream>>>(attn_out, ln2g, ln2b, nullptr, h2b);

  // FFN
  gemm(h2b, w1b, b1, nullptr, nullptr, fbuf, M_, FF_, D_, 1);
  gemm(fbuf, w2b, b2, attn_out, out, nullptr, M_, D_, FF_, 0);
}
